// FactGCN2_26328149525018
// MI455X (gfx1250) — compile-verified, matrix operand order checked by reading, not run on hardware
//
#include <hip/hip_runtime.h>
#include <hip/hip_bf16.h>

// ---------------------------------------------------------------------------
// Types for CDNA5 WMMA (wave32): v_wmma_f32_16x16x32_bf16
// ---------------------------------------------------------------------------
typedef __bf16 bf16;
typedef __attribute__((ext_vector_type(16))) __bf16 v16bf;
typedef __attribute__((ext_vector_type(8)))  __bf16 v8bf;
typedef __attribute__((ext_vector_type(8)))  float  v8f;
typedef __attribute__((ext_vector_type(4)))  int    v4i;

#define CDIV(a, b) (((a) + (b) - 1) / (b))

// ---- CDNA5 async global->LDS copy (ASYNCcnt path), with safe fallback ----
// Probe result: __builtin_amdgcn_global_load_async_to_lds_b128 expects
// (v4i addrspace(1)*, v4i addrspace(3)*, imm offset, imm cpol).
#if defined(__gfx1250__) && __has_builtin(__builtin_amdgcn_global_load_async_to_lds_b128)
  #define HAVE_ASYNC_LDS 1
  typedef __attribute__((address_space(1))) v4i* as1_v4i;
  typedef __attribute__((address_space(3))) v4i* as3_v4i;
  #define ASYNC_CP_B128(gsrc, ldst)                                          \
    __builtin_amdgcn_global_load_async_to_lds_b128(                          \
        (as1_v4i)(gsrc), (as3_v4i)(ldst), 0, 0)
  #if __has_builtin(__builtin_amdgcn_s_wait_asynccnt)
    #define ASYNC_WAIT() __builtin_amdgcn_s_wait_asynccnt(0)
  #else
    #define ASYNC_WAIT() asm volatile("s_wait_asynccnt 0" ::: "memory")
  #endif
#else
  #define HAVE_ASYNC_LDS 0
  #define ASYNC_CP_B128(gsrc, ldst)                                          \
    (*(v8bf*)(ldst) = *(const v8bf*)(gsrc))
  #define ASYNC_WAIT() ((void)0)
#endif

// ---------------------------------------------------------------------------
// Fragment load, 16-bit A/B striping for 16x16x32 (ISA 7.12.2):
//   lanes 0-15 : elems 0-7 -> K=k..k+7,    elems 8-15 -> K=k+16..k+23
//   lanes 16-31: elems 0-7 -> K=k+8..k+15, elems 8-15 -> K=k+24..k+31
// ---------------------------------------------------------------------------
__device__ __forceinline__ v16bf load_frag(const bf16* __restrict__ rowbase,
                                           int k, int kb) {
  v16bf f;
  *((v8bf*)&f)       = *(const v8bf*)(rowbase + k + kb);
  *(((v8bf*)&f) + 1) = *(const v8bf*)(rowbase + k + kb + 16);
  return f;
}

// fragment from an LDS tile row (row pitch 80B; halves 32B apart)
__device__ __forceinline__ v16bf lds_frag(const char* base) {
  v16bf f;
  *((v8bf*)&f)       = *(const v8bf*)(base);
  *(((v8bf*)&f) + 1) = *(const v8bf*)(base + 32);
  return f;
}

__device__ __forceinline__ void epilogue(v8f acc, int rbase, int col, int M, int N,
                                         const float* __restrict__ bias,
                                         const float* __restrict__ rowScale, int act,
                                         float* __restrict__ Cf, int ldc, int coff,
                                         bf16* __restrict__ Cb, int ldcb, int coffb) {
  if (col >= N) return;
  float bv = bias ? bias[col] : 0.f;
#pragma unroll
  for (int r = 0; r < 8; ++r) {
    int row = rbase + r;
    if (row < M) {
      float v = acc[r] + bv;
      if (act == 1) v = fmaxf(v, 0.f);
      if (rowScale) v *= rowScale[row];
      if (Cf) Cf[(size_t)row * ldc + coff + col] = v;
      if (Cb) Cb[(size_t)row * ldcb + coffb + col] = (bf16)v;
    }
  }
}

// ---------------------------------------------------------------------------
// 1-wave GEMM (small M): C[M,N] = A[M,K] @ Bt[N,K]^T; 16x64 tile per wave.
// ---------------------------------------------------------------------------
__global__ void __launch_bounds__(32)
gemm_bf16_wmma(const bf16* __restrict__ A, int lda,
               const bf16* __restrict__ Bt, int ldb,
               const float* __restrict__ bias,
               const float* __restrict__ rowScale, int act,
               float* __restrict__ Cf, int ldc, int coff,
               bf16* __restrict__ Cb, int ldcb, int coffb,
               int M, int N, int K) {
  const int lane = threadIdx.x & 31;
  const int tN = blockIdx.x * 64;
  const int tM = blockIdx.y * 16;
  const int mrow = tM + (lane & 15);
  const bf16* arow = A + (size_t)((mrow < M) ? mrow : 0) * lda;
  const int kb = (lane & 16) ? 8 : 0;
  const int ncol = lane & 15;
  const int n0 = tN + ncol;
  const bf16* b0r = Bt + (size_t)((n0      < N) ? n0      : N - 1) * ldb;
  const bf16* b1r = Bt + (size_t)((n0 + 16 < N) ? n0 + 16 : N - 1) * ldb;
  const bf16* b2r = Bt + (size_t)((n0 + 32 < N) ? n0 + 32 : N - 1) * ldb;
  const bf16* b3r = Bt + (size_t)((n0 + 48 < N) ? n0 + 48 : N - 1) * ldb;

  v8f acc0 = {}, acc1 = {}, acc2 = {}, acc3 = {};
  for (int k = 0; k < K; k += 32) {
    v16bf a  = load_frag(arow, k, kb);
    v16bf b0 = load_frag(b0r, k, kb);
    v16bf b1 = load_frag(b1r, k, kb);
    v16bf b2 = load_frag(b2r, k, kb);
    v16bf b3 = load_frag(b3r, k, kb);
    acc0 = __builtin_amdgcn_wmma_f32_16x16x32_bf16(false, a, false, b0, (short)0, acc0, false, false);
    acc1 = __builtin_amdgcn_wmma_f32_16x16x32_bf16(false, a, false, b1, (short)0, acc1, false, false);
    acc2 = __builtin_amdgcn_wmma_f32_16x16x32_bf16(false, a, false, b2, (short)0, acc2, false, false);
    acc3 = __builtin_amdgcn_wmma_f32_16x16x32_bf16(false, a, false, b3, (short)0, acc3, false, false);
  }
  const int rbase = tM + ((lane & 16) ? 8 : 0);
  epilogue(acc0, rbase, tN + ncol,      M, N, bias, rowScale, act, Cf, ldc, coff, Cb, ldcb, coffb);
  epilogue(acc1, rbase, tN + ncol + 16, M, N, bias, rowScale, act, Cf, ldc, coff, Cb, ldcb, coffb);
  epilogue(acc2, rbase, tN + ncol + 32, M, N, bias, rowScale, act, Cf, ldc, coff, Cb, ldcb, coffb);
  epilogue(acc3, rbase, tN + ncol + 48, M, N, bias, rowScale, act, Cf, ldc, coff, Cb, ldcb, coffb);
}

// ---------------------------------------------------------------------------
// 8-wave LDS-staged GEMM (M % 128 == 0): block computes 128(M) x 64(N).
// The shared B tile (64 cols x 32 K = 4KB) is staged into LDS once per block
// per K-step via CDNA5 async global->LDS B128 copies (double buffered,
// s_wait_asynccnt + barrier), then all 8 waves read fragments with ds loads.
// LDS row pitch 80B keeps the 16-lane b128 reads conflict-free.
// ---------------------------------------------------------------------------
__global__ void __launch_bounds__(256)
gemm_bf16_wmma_lds(const bf16* __restrict__ A, int lda,
                   const bf16* __restrict__ Bt, int ldb,
                   const float* __restrict__ bias,
                   const float* __restrict__ rowScale, int act,
                   float* __restrict__ Cf, int ldc, int coff,
                   bf16* __restrict__ Cb, int ldcb, int coffb,
                   int M, int N, int K) {
  __shared__ __align__(16) char ldsB[2][64 * 80];
  const int tid  = threadIdx.x;
  const int lane = tid & 31;
  const int wv   = tid >> 5;                 // 0..7 -> M sub-tile
  const int tN = blockIdx.x * 64;
  const int tM = blockIdx.y * 128 + wv * 16;

  // --- B staging addresses: thread copies 16B; row = tid/4, chunk = tid%4 ---
  const int srow   = tid >> 2;               // 0..63 (column within N tile)
  const int schunk = (tid & 3) * 16;         // byte chunk within 64B K-segment
  int sn = tN + srow; if (sn >= N) sn = N - 1;
  const char* sbase = (const char*)(Bt + (size_t)sn * ldb) + schunk;
  char* dst0 = &ldsB[0][srow * 80 + schunk];
  char* dst1 = &ldsB[1][srow * 80 + schunk];

  // --- per-wave A row / fragment geometry ---
  const int mrow = tM + (lane & 15);
  const bf16* arow = A + (size_t)((mrow < M) ? mrow : 0) * lda;
  const int kb   = (lane & 16) ? 8 : 0;
  const int ncol = lane & 15;
  const int kboff = kb * 2;                  // byte offset of frag within row

  // prologue: stage K-step 0 into buffer 0
  ASYNC_CP_B128(sbase, dst0);
  ASYNC_WAIT();
  __syncthreads();

  v8f acc0 = {}, acc1 = {}, acc2 = {}, acc3 = {};
  int buf = 0;
  for (int k = 0; k < K; k += 32) {
    if (k + 32 < K)  // prefetch next K-step into the other buffer
      ASYNC_CP_B128(sbase + (size_t)(k + 32) * 2, buf ? dst0 : dst1);
    const char* lb = ldsB[buf];
    v16bf a  = load_frag(arow, k, kb);
    v16bf b0 = lds_frag(lb + (ncol     ) * 80 + kboff);
    v16bf b1 = lds_frag(lb + (ncol + 16) * 80 + kboff);
    v16bf b2 = lds_frag(lb + (ncol + 32) * 80 + kboff);
    v16bf b3 = lds_frag(lb + (ncol + 48) * 80 + kboff);
    acc0 = __builtin_amdgcn_wmma_f32_16x16x32_bf16(false, a, false, b0, (short)0, acc0, false, false);
    acc1 = __builtin_amdgcn_wmma_f32_16x16x32_bf16(false, a, false, b1, (short)0, acc1, false, false);
    acc2 = __builtin_amdgcn_wmma_f32_16x16x32_bf16(false, a, false, b2, (short)0, acc2, false, false);
    acc3 = __builtin_amdgcn_wmma_f32_16x16x32_bf16(false, a, false, b3, (short)0, acc3, false, false);
    ASYNC_WAIT();
    __syncthreads();
    buf ^= 1;
  }
  const int rbase = tM + ((lane & 16) ? 8 : 0);
  epilogue(acc0, rbase, tN + ncol,      M, N, bias, rowScale, act, Cf, ldc, coff, Cb, ldcb, coffb);
  epilogue(acc1, rbase, tN + ncol + 16, M, N, bias, rowScale, act, Cf, ldc, coff, Cb, ldcb, coffb);
  epilogue(acc2, rbase, tN + ncol + 32, M, N, bias, rowScale, act, Cf, ldc, coff, Cb, ldcb, coffb);
  epilogue(acc3, rbase, tN + ncol + 48, M, N, bias, rowScale, act, Cf, ldc, coff, Cb, ldcb, coffb);
}

// ---------------------------------------------------------------------------
// Weight prep: Wt[n, k] = bf16(W[k, n]), K zero-padded to Kpad.
// ---------------------------------------------------------------------------
__global__ void wt_prep_k(const float* __restrict__ W, bf16* __restrict__ Wt,
                          int K, int N, int Kpad) {
  int i = blockIdx.x * 256 + threadIdx.x;
  if (i < N * Kpad) {
    int n = i / Kpad, k = i % Kpad;
    Wt[i] = (k < K) ? (bf16)W[(size_t)k * N + n] : (bf16)0.f;
  }
}

__global__ void cvt_bf16_k(const float* __restrict__ in, int ldi,
                           bf16* __restrict__ out, int ldo, int rows, int cols) {
  int i = blockIdx.x * 256 + threadIdx.x;
  if (i < rows * cols) {
    int r = i / cols, c = i % cols;
    out[(size_t)r * ldo + c] = (bf16)in[(size_t)r * ldi + c];
  }
}

__global__ void cvtT_bf16_k(const float* __restrict__ in, int ldi,
                            bf16* __restrict__ out, int ldo, int rows, int cols) {
  int i = blockIdx.x * 256 + threadIdx.x;
  if (i < rows * cols) {
    int r = i / cols, c = i % cols;
    out[(size_t)c * ldo + r] = (bf16)in[(size_t)r * ldi + c];
  }
}

// ---------------------------------------------------------------------------
// Additive-tanh attention + softmax over NX keys (one block per fact row).
// ---------------------------------------------------------------------------
__global__ void attn_k(const float* __restrict__ pf, const float* __restrict__ px,
                       const float* __restrict__ w_ap, const float* __restrict__ b_ap,
                       int NX, bf16* __restrict__ a_bf, int lda) {
  __shared__ float sc[64];
  __shared__ float prob[64];
  const int f = blockIdx.x;
  const int tid = threadIdx.x;
  const int lane = tid & 31, wv = tid >> 5;
  const float* pfr = pf + (size_t)f * 512;
  for (int i = wv; i < NX; i += 8) {
    const float* pxr = px + (size_t)i * 512;
    float s = 0.f;
    for (int d = lane; d < 512; d += 32) s += tanhf(pfr[d] + pxr[d]) * w_ap[d];
    for (int off = 16; off; off >>= 1) s += __shfl_xor(s, off, 32);
    if (lane == 0) sc[i] = s + b_ap[0];
  }
  __syncthreads();
  if (tid == 0) {
    float mx = -1e30f;
    for (int i = 0; i < NX; ++i) mx = fmaxf(mx, sc[i]);
    float sum = 0.f;
    for (int i = 0; i < NX; ++i) { float e = __expf(sc[i] - mx); prob[i] = e; sum += e; }
    float inv = 1.f / sum;
    for (int i = 0; i < NX; ++i) prob[i] *= inv;
  }
  __syncthreads();
  if (tid < NX) a_bf[(size_t)f * lda + tid] = (bf16)prob[tid];
}

// ---------------------------------------------------------------------------
// Message passing: deterministic gather-mean over incoming edges.
// ---------------------------------------------------------------------------
__global__ void mp_k(const float* __restrict__ h, const int* __restrict__ src,
                     const int* __restrict__ dst, int E,
                     bf16* __restrict__ concat_bf, int ldcat,
                     float* __restrict__ deg) {
  __shared__ int cnt_s[257];
  __shared__ int matches[6144];
  const int f = blockIdx.x;
  const int tid = threadIdx.x;
  int c = 0;
  for (int e = tid; e < E; e += 256) c += (dst[e] == f);
  cnt_s[tid] = c;
  __syncthreads();
  if (tid == 0) {
    int acc = 0;
    for (int t = 0; t < 256; ++t) { int v = cnt_s[t]; cnt_s[t] = acc; acc += v; }
    cnt_s[256] = acc;
  }
  __syncthreads();
  int pos = cnt_s[tid];
  for (int e = tid; e < E; e += 256)
    if (dst[e] == f) matches[pos++] = src[e];
  __syncthreads();
  const int total = cnt_s[256];
  const float inv = 1.f / fmaxf((float)total, 1.f);
  for (int d = tid; d < 512; d += 256) {
    float acc = 0.f;
    for (int m = 0; m < total; ++m) acc += h[(size_t)matches[m] * 512 + d];
    concat_bf[(size_t)f * ldcat + d] = (bf16)(acc * inv);
  }
  if (tid == 0) deg[f] = (float)total;
}

__global__ void gatemul_k(const float* __restrict__ gp, const float* __restrict__ cat3f,
                          bf16* __restrict__ gated, int n) {
  int i = blockIdx.x * 256 + threadIdx.x;
  if (i < n) {
    float g = 1.f / (1.f + __expf(-gp[i]));
    gated[i] = (bf16)(g * cat3f[i]);
  }
}

__global__ void final_k(const float* __restrict__ hout, const float* __restrict__ h,
                        const float* __restrict__ deg, float* __restrict__ out,
                        int D, int n) {
  int i = blockIdx.x * 256 + threadIdx.x;
  if (i < n) {
    int f = i / D;
    out[i] = (deg[f] > 0.f) ? hout[i] : h[i];
  }
}

// ---------------------------------------------------------------------------
// Host-side orchestration
// ---------------------------------------------------------------------------
extern "C" void kernel_launch(void* const* d_in, const int* in_sizes, int n_in,
                              void* d_out, int out_size, void* d_ws, size_t ws_size,
                              hipStream_t stream) {
  (void)in_sizes; (void)n_in; (void)out_size; (void)ws_size;
  const int B = 8, NF = 384, NI = 36, NS = 48, E = 6144;
  const int IN_D = 512, IMG_D = 2048, SEM_D = 300, SEM_P = 320;
  const int G3 = 1536;

  const float* fact_h = (const float*)d_in[0];
  const float* img_h  = (const float*)d_in[1];
  const float* sem_h  = (const float*)d_in[2];
  const float* attp   = (const float*)d_in[3];
  const int*   e_src  = (const int*)d_in[4];
  const int*   e_dst  = (const int*)d_in[5];
  const float *W_cifp = (const float*)d_in[6],  *b_cifp = (const float*)d_in[7];
  const float *W_ciip = (const float*)d_in[8],  *b_ciip = (const float*)d_in[9];
  const float *W_iap  = (const float*)d_in[10], *b_iap  = (const float*)d_in[11];
  const float *W_csfp = (const float*)d_in[12], *b_csfp = (const float*)d_in[13];
  const float *W_csnp = (const float*)d_in[14], *b_csnp = (const float*)d_in[15];
  const float *W_sap  = (const float*)d_in[16], *b_sap  = (const float*)d_in[17];
  const float *W_node = (const float*)d_in[18], *b_node = (const float*)d_in[19];
  const float *W_appl = (const float*)d_in[20], *b_appl = (const float*)d_in[21];
  const float *W_ig   = (const float*)d_in[22], *b_ig   = (const float*)d_in[23];
  const float *W_sg   = (const float*)d_in[24], *b_sg   = (const float*)d_in[25];
  const float *W_fg   = (const float*)d_in[26], *b_fg   = (const float*)d_in[27];
  const float *W_gate = (const float*)d_in[28], *b_gate = (const float*)d_in[29];
  const float *W_out  = (const float*)d_in[30], *b_out  = (const float*)d_in[31];

  char* p = (char*)d_ws;
  auto alloc = [&](size_t bytes) -> void* {
    void* r = (void*)p; p += (bytes + 255) & ~(size_t)255; return r;
  };
  bf16* Wt_cifp = (bf16*)alloc((size_t)512 * 512 * 2);
  bf16* Wt_ciip = (bf16*)alloc((size_t)512 * 2048 * 2);
  bf16* Wt_csfp = (bf16*)alloc((size_t)512 * 512 * 2);
  bf16* Wt_csnp = (bf16*)alloc((size_t)512 * SEM_P * 2);
  bf16* Wt_node = (bf16*)alloc((size_t)512 * 512 * 2);
  bf16* Wt_appl = (bf16*)alloc((size_t)512 * 1024 * 2);
  bf16* Wt_ig   = (bf16*)alloc((size_t)512 * 2048 * 2);
  bf16* Wt_sg   = (bf16*)alloc((size_t)512 * SEM_P * 2);
  bf16* Wt_fg   = (bf16*)alloc((size_t)512 * 512 * 2);
  bf16* Wt_gate = (bf16*)alloc((size_t)G3 * G3 * 2);
  bf16* Wt_out  = (bf16*)alloc((size_t)512 * G3 * 2);
  bf16* fh_bf   = (bf16*)alloc((size_t)NF * 512 * 2);
  bf16* ih_bf   = (bf16*)alloc((size_t)NI * 2048 * 2);
  bf16* ihT_bf  = (bf16*)alloc((size_t)2048 * 64 * 2);
  bf16* sh_bf   = (bf16*)alloc((size_t)NS * SEM_P * 2);
  bf16* shT_bf  = (bf16*)alloc((size_t)SEM_P * 64 * 2);
  bf16* aimg_bf = (bf16*)alloc((size_t)NF * 64 * 2);
  bf16* asem_bf = (bf16*)alloc((size_t)NF * 64 * 2);
  bf16* img_bf  = (bf16*)alloc((size_t)NF * 2048 * 2);
  bf16* sem_bf  = (bf16*)alloc((size_t)NF * SEM_P * 2);
  bf16* cat_bf  = (bf16*)alloc((size_t)NF * 1024 * 2);   // [neigh | h]
  bf16* hf_bf   = (bf16*)alloc((size_t)NF * 512 * 2);
  bf16* cat3_bf = (bf16*)alloc((size_t)NF * G3 * 2);
  bf16* gatd_bf = (bf16*)alloc((size_t)NF * G3 * 2);
  float* npjf   = (float*)alloc((size_t)NF * 512 * 4);
  float* nspjf  = (float*)alloc((size_t)NF * 512 * 4);
  float* hfp32  = (float*)alloc((size_t)NF * 512 * 4);
  float* ipjf   = (float*)alloc((size_t)48 * 512 * 4);
  float* spjf   = (float*)alloc((size_t)48 * 512 * 4);
  float* cat3f  = (float*)alloc((size_t)NF * G3 * 4);
  float* gatepf = (float*)alloc((size_t)NF * G3 * 4);
  float* houtf  = (float*)alloc((size_t)NF * 512 * 4);
  float* degf   = (float*)alloc((size_t)NF * 4);
  const size_t used = (size_t)(p - (char*)d_ws);

  (void)hipMemsetAsync(d_ws, 0, used, stream);

  auto ewise = [&](int n) { return dim3(CDIV(n, 256)); };
  wt_prep_k<<<ewise(512 * 512),   256, 0, stream>>>(W_cifp, Wt_cifp, 512,  512, 512);
  wt_prep_k<<<ewise(512 * 2048),  256, 0, stream>>>(W_ciip, Wt_ciip, 2048, 512, 2048);
  wt_prep_k<<<ewise(512 * 512),   256, 0, stream>>>(W_csfp, Wt_csfp, 512,  512, 512);
  wt_prep_k<<<ewise(512 * SEM_P), 256, 0, stream>>>(W_csnp, Wt_csnp, SEM_D, 512, SEM_P);
  wt_prep_k<<<ewise(512 * 512),   256, 0, stream>>>(W_node, Wt_node, 512,  512, 512);
  wt_prep_k<<<ewise(512 * 1024),  256, 0, stream>>>(W_appl, Wt_appl, 1024, 512, 1024);
  wt_prep_k<<<ewise(512 * 2048),  256, 0, stream>>>(W_ig,   Wt_ig,   2048, 512, 2048);
  wt_prep_k<<<ewise(512 * SEM_P), 256, 0, stream>>>(W_sg,   Wt_sg,   SEM_D, 512, SEM_P);
  wt_prep_k<<<ewise(512 * 512),   256, 0, stream>>>(W_fg,   Wt_fg,   512,  512, 512);
  wt_prep_k<<<ewise(G3 * G3),     256, 0, stream>>>(W_gate, Wt_gate, G3,   G3,  G3);
  wt_prep_k<<<ewise(512 * G3),    256, 0, stream>>>(W_out,  Wt_out,  G3,   512, G3);

  auto gemm = [&](const bf16* A, int lda, const bf16* Bt, int ldb,
                  const float* bias, const float* rs, int act,
                  float* Cf, int ldc, int coff, bf16* Cb, int ldcb, int coffb,
                  int M, int N, int K) {
    if ((M & 127) == 0) {
      dim3 g(CDIV(N, 64), M / 128);
      gemm_bf16_wmma_lds<<<g, 256, 0, stream>>>(A, lda, Bt, ldb, bias, rs, act,
                                                Cf, ldc, coff, Cb, ldcb, coffb, M, N, K);
    } else {
      dim3 g(CDIV(N, 64), CDIV(M, 16));
      gemm_bf16_wmma<<<g, 32, 0, stream>>>(A, lda, Bt, ldb, bias, rs, act,
                                           Cf, ldc, coff, Cb, ldcb, coffb, M, N, K);
    }
  };

  for (int b = 0; b < B; ++b) {
    const float* fh = fact_h + (size_t)b * NF * IN_D;
    const float* ih = img_h  + (size_t)b * NI * IMG_D;
    const float* sh = sem_h  + (size_t)b * NS * SEM_D;
    const float* at = attp   + (size_t)b * NF;
    const int* sb = e_src + (size_t)b * E;
    const int* db = e_dst + (size_t)b * E;
    float* outb = (float*)d_out + (size_t)b * NF * 512;

    cvt_bf16_k <<<ewise(NF * 512),   256, 0, stream>>>(fh, 512,   fh_bf, 512,   NF, 512);
    cvt_bf16_k <<<ewise(NI * 2048),  256, 0, stream>>>(ih, 2048,  ih_bf, 2048,  NI, 2048);
    cvtT_bf16_k<<<ewise(NI * 2048),  256, 0, stream>>>(ih, 2048,  ihT_bf, 64,   NI, 2048);
    cvt_bf16_k <<<ewise(NS * SEM_D), 256, 0, stream>>>(sh, SEM_D, sh_bf, SEM_P, NS, SEM_D);
    cvtT_bf16_k<<<ewise(NS * SEM_D), 256, 0, stream>>>(sh, SEM_D, shT_bf, 64,   NS, SEM_D);

    gemm(fh_bf, 512,  Wt_cifp, 512,  b_cifp, nullptr, 0, npjf,  512, 0, nullptr, 0, 0,    NF, 512, 512);
    gemm(ih_bf, 2048, Wt_ciip, 2048, b_ciip, nullptr, 0, ipjf,  512, 0, nullptr, 0, 0,    NI, 512, 2048);
    gemm(fh_bf, 512,  Wt_csfp, 512,  b_csfp, nullptr, 0, nspjf, 512, 0, nullptr, 0, 0,    NF, 512, 512);
    gemm(sh_bf, SEM_P, Wt_csnp, SEM_P, b_csnp, nullptr, 0, spjf, 512, 0, nullptr, 0, 0,   NS, 512, SEM_P);
    gemm(fh_bf, 512,  Wt_node, 512,  b_node, nullptr, 0, hfp32, 512, 0, cat_bf, 1024, 512, NF, 512, 512);

    attn_k<<<NF, 256, 0, stream>>>(npjf,  ipjf, W_iap, b_iap, NI, aimg_bf, 64);
    attn_k<<<NF, 256, 0, stream>>>(nspjf, spjf, W_sap, b_sap, NS, asem_bf, 64);

    gemm(aimg_bf, 64, ihT_bf, 64, nullptr, nullptr, 0, nullptr, 0, 0, img_bf, 2048, 0, NF, 2048, 64);
    gemm(asem_bf, 64, shT_bf, 64, nullptr, nullptr, 0, nullptr, 0, 0, sem_bf, SEM_P, 0, NF, SEM_D, 64);

    mp_k<<<NF, 256, 0, stream>>>(hfp32, sb, db, E, cat_bf, 1024, degf);
    gemm(cat_bf, 1024, Wt_appl, 1024, b_appl, at, 1, nullptr, 0, 0, hf_bf, 512, 0, NF, 512, 1024);

    gemm(hf_bf,  512,   Wt_fg, 512,   b_fg, nullptr, 0, cat3f, G3, 0,    cat3_bf, G3, 0,    NF, 512, 512);
    gemm(img_bf, 2048,  Wt_ig, 2048,  b_ig, nullptr, 0, cat3f, G3, 512,  cat3_bf, G3, 512,  NF, 512, 2048);
    gemm(sem_bf, SEM_P, Wt_sg, SEM_P, b_sg, nullptr, 0, cat3f, G3, 1024, cat3_bf, G3, 1024, NF, 512, SEM_P);

    gemm(cat3_bf, G3, Wt_gate, G3, b_gate, nullptr, 0, gatepf, G3, 0, nullptr, 0, 0, NF, G3, G3);
    gatemul_k<<<ewise(NF * G3), 256, 0, stream>>>(gatepf, cat3f, gatd_bf, NF * G3);
    gemm(gatd_bf, G3, Wt_out, G3, b_out, nullptr, 0, houtf, 512, 0, nullptr, 0, 0, NF, 512, G3);
    final_k<<<ewise(NF * 512), 256, 0, stream>>>(houtf, hfp32, degf, outb, 512, NF * 512);
  }
}
